// ScoreNet_69827578298882
// MI455X (gfx1250) — matrix operand order repaired and checked by reading, not compiled, not run on hardware
//
#include <hip/hip_runtime.h>

typedef __attribute__((ext_vector_type(16))) __bf16 v16bf;
typedef __attribute__((ext_vector_type(8)))  float  v8f;

union FragU { uint4 q[2]; v16bf v; };

static __device__ __forceinline__ unsigned short f2bf(float f) {
  unsigned u = __float_as_uint(f);
  u += 0x7fffu + ((u >> 16) & 1u);   // round-to-nearest-even
  return (unsigned short)(u >> 16);
}

static __device__ __forceinline__ float swishf(float x) {
  return x * (1.0f / (1.0f + __expf(-x)));
}

// ---------------------------------------------------------------------------
// Prep kernel: time embedding (B x 80), bf16 weight conversion, prefix sums,
// folded final projection Wc = lb_W @ la_W  (3x32), bc = lb_W@la_b + lb_b.
// ---------------------------------------------------------------------------
__global__ __launch_bounds__(256) void prep_kernel(
    const float* __restrict__ t, const float* __restrict__ Wf,
    const float* __restrict__ embW, const float* __restrict__ embB,
    const float* __restrict__ l0W, const float* __restrict__ l1W,
    const float* __restrict__ l2W,
    const float* __restrict__ laW, const float* __restrict__ laB,
    const float* __restrict__ lbW, const float* __restrict__ lbB,
    const int* __restrict__ num_paths,
    float* __restrict__ wsEmb, unsigned short* __restrict__ wsWbf,
    int* __restrict__ wsPrefix, float* __restrict__ wsWc,
    float* __restrict__ wsBc, int B)
{
  const int tid = threadIdx.x;

  // per-graph time embedding: sin/cos Fourier (80) -> 80x80 linear -> swish
  if (tid < B) {
    float tv = t[tid];
    float e[80];
    #pragma unroll
    for (int i = 0; i < 40; ++i) {
      float xp = tv * Wf[i] * 6.283185307179586f;
      e[i]      = __sinf(xp);
      e[i + 40] = __cosf(xp);
    }
    for (int c = 0; c < 80; ++c) {
      float acc = embB[c];
      const float* wr = embW + c * 80;
      #pragma unroll 4
      for (int i = 0; i < 80; ++i) acc += e[i] * wr[i];
      wsEmb[tid * 80 + c] = swishf(acc);
    }
  }

  // convert hidden-layer weights to bf16, packed contiguously: W0 | W1 | W2
  for (int i = tid; i < 43008; i += blockDim.x) {
    float w;
    if (i < 32768)      w = l0W[i];
    else if (i < 40960) w = l1W[i - 32768];
    else                w = l2W[i - 40960];
    wsWbf[i] = f2bf(w);
  }

  if (tid == 0) {
    int s = 0;
    for (int b = 0; b < B; ++b) { wsPrefix[b] = s; s += num_paths[b]; }
    wsPrefix[B] = s;
    for (int j = 0; j < 3; ++j) {
      float bb = lbB[j];
      for (int m = 0; m < 3; ++m) bb += lbW[j * 3 + m] * laB[m];
      wsBc[j] = bb;
      for (int k = 0; k < 32; ++k) {
        float w = 0.f;
        for (int m = 0; m < 3; ++m) w += lbW[j * 3 + m] * laW[m * 32 + k];
        wsWc[j * 32 + k] = w;
      }
    }
  }
}

// ---------------------------------------------------------------------------
// Main fused kernel: gather + up-projection -> A[128x256] bf16 in LDS,
// then 3 WMMA GEMM layers with swish, then folded 32->3 epilogue.
// One workgroup = 128 path rows, 8 waves x 16 rows each.
// ---------------------------------------------------------------------------
__global__ __launch_bounds__(256) void mlp_kernel(
    const float* __restrict__ node_attr, const float* __restrict__ perturb,
    const int* __restrict__ paths,
    const float* __restrict__ upW, const float* __restrict__ upB,
    const float* __restrict__ b0, const float* __restrict__ b1,
    const float* __restrict__ b2,
    const float* __restrict__ wsEmb, const unsigned short* __restrict__ wsWbf,
    const int* __restrict__ wsPrefix, const float* __restrict__ wsWc,
    const float* __restrict__ wsBc,
    float* __restrict__ out, int P, int B)
{
  extern __shared__ char smem[];
  unsigned short* sW0  = (unsigned short*)smem;   // 128*256 bf16 = 64 KB
  unsigned short* sW1  = sW0 + 128 * 256;         //  64*128 bf16 = 16 KB
  unsigned short* sW2  = sW1 + 64 * 128;          //  32*64  bf16 =  4 KB
  unsigned short* sAct = sW2 + 32 * 64;           // 128*256 bf16 = 64 KB (A tile)
  unsigned short* sH1  = sAct;                    // 128*128 bf16 (reuses A)
  unsigned short* sH2  = sAct + 16384;            // 128*64  bf16
  float*          sH3  = (float*)(sAct + 24576);  // 128*32  f32

  const int tid  = threadIdx.x;
  const int lane = tid & 31;
  const int wave = tid >> 5;
  const int p0   = blockIdx.x * 128;

  // ---- stage bf16 weights into LDS (contiguous copy, b128 granularity) ----
  {
    const uint4* src = (const uint4*)wsWbf;
    uint4* dst = (uint4*)sW0;
    #pragma unroll
    for (int i = tid; i < 43008 / 8; i += 256) dst[i] = src[i];
  }

  // ---- build A tile: concat(g0,g1,g2,g3,emb) + perturb @ up_W^T + up_b ----
  {
    const int row = tid >> 1;
    const int c0  = (tid & 1) * 128;
    const int p   = p0 + row;
    // graph id via binary search over prefix sums
    int lo = 0, hi = B - 1;
    while (lo < hi) {
      int mid = (lo + hi + 1) >> 1;
      if (wsPrefix[mid] <= p) lo = mid; else hi = mid - 1;
    }
    const float* embR = wsEmb + lo * 80;
    int idx[4];
    #pragma unroll
    for (int j = 0; j < 4; ++j) idx[j] = paths[j * P + p];
    const float px = perturb[p * 3 + 0];
    const float py = perturb[p * 3 + 1];
    const float pz = perturb[p * 3 + 2];
    unsigned short* aRow = sAct + row * 256;
    for (int c = c0; c < c0 + 128; c += 2) {
      float v[2];
      #pragma unroll
      for (int u = 0; u < 2; ++u) {
        int cc = c + u;
        float x;
        if (cc < 176) {
          int j = cc / 44;
          x = node_attr[(long)idx[j] * 44 + (cc - j * 44)];
        } else {
          x = embR[cc - 176];
        }
        x += upB[cc] + px * upW[cc * 3] + py * upW[cc * 3 + 1] + pz * upW[cc * 3 + 2];
        v[u] = x;
      }
      unsigned pack = (unsigned)f2bf(v[0]) | ((unsigned)f2bf(v[1]) << 16);
      *(unsigned*)(aRow + c) = pack;
    }
  }
  __syncthreads();

  // fragment addressing (per ISA 16-bit A/B VGPR layouts)
  const int m0      = wave * 16;
  const int aM      = m0 + (lane & 15);
  const int aKo     = ((lane >> 4) & 1) * 8;    // A: lane halves K {0..7,16..23}/{8..15,24..31}
  const int bN15    = lane & 15;
  const int bKo     = ((lane >> 4) & 1) * 8;    // B: lane halves K {0..7,16..23}/{8..15,24..31}
  const int rowBase = m0 + ((lane >> 4) & 1) * 8;
  const int col     = lane & 15;

  // ---- GEMM1: [16x256] x [256->128], swish ----
  v8f acc1[8] = {};
  for (int kk = 0; kk < 256; kk += 32) {
    FragU a;
    a.q[0] = *(const uint4*)(sAct + aM * 256 + kk + aKo);
    a.q[1] = *(const uint4*)(sAct + aM * 256 + kk + aKo + 16);
    #pragma unroll
    for (int nt = 0; nt < 8; ++nt) {
      FragU b;
      const uint4* bp = (const uint4*)(sW0 + (nt * 16 + bN15) * 256 + kk + bKo);
      b.q[0] = bp[0]; b.q[1] = bp[2];
      acc1[nt] = __builtin_amdgcn_wmma_f32_16x16x32_bf16(
          false, a.v, false, b.v, (short)0, acc1[nt], false, false);
    }
  }
  __syncthreads();   // all waves done reading A before h1 overwrites it
  #pragma unroll
  for (int nt = 0; nt < 8; ++nt) {
    int n = nt * 16 + col;
    float bias = b0[n];
    #pragma unroll
    for (int r = 0; r < 8; ++r) {
      float x = acc1[nt][r] + bias;
      sH1[(rowBase + r) * 128 + n] = f2bf(swishf(x));
    }
  }

  // ---- GEMM2: [16x128] x [128->64], swish (own rows, same-wave LDS order) ----
  v8f acc2[4] = {};
  for (int kk = 0; kk < 128; kk += 32) {
    FragU a;
    a.q[0] = *(const uint4*)(sH1 + aM * 128 + kk + aKo);
    a.q[1] = *(const uint4*)(sH1 + aM * 128 + kk + aKo + 16);
    #pragma unroll
    for (int nt = 0; nt < 4; ++nt) {
      FragU b;
      const uint4* bp = (const uint4*)(sW1 + (nt * 16 + bN15) * 128 + kk + bKo);
      b.q[0] = bp[0]; b.q[1] = bp[2];
      acc2[nt] = __builtin_amdgcn_wmma_f32_16x16x32_bf16(
          false, a.v, false, b.v, (short)0, acc2[nt], false, false);
    }
  }
  #pragma unroll
  for (int nt = 0; nt < 4; ++nt) {
    int n = nt * 16 + col;
    float bias = b1[n];
    #pragma unroll
    for (int r = 0; r < 8; ++r) {
      float x = acc2[nt][r] + bias;
      sH2[(rowBase + r) * 64 + n] = f2bf(swishf(x));
    }
  }

  // ---- GEMM3: [16x64] x [64->32], swish -> f32 LDS ----
  v8f acc3[2] = {};
  #pragma unroll
  for (int kk = 0; kk < 64; kk += 32) {
    FragU a;
    a.q[0] = *(const uint4*)(sH2 + aM * 64 + kk + aKo);
    a.q[1] = *(const uint4*)(sH2 + aM * 64 + kk + aKo + 16);
    #pragma unroll
    for (int nt = 0; nt < 2; ++nt) {
      FragU b;
      const uint4* bp = (const uint4*)(sW2 + (nt * 16 + bN15) * 64 + kk + bKo);
      b.q[0] = bp[0]; b.q[1] = bp[2];
      acc3[nt] = __builtin_amdgcn_wmma_f32_16x16x32_bf16(
          false, a.v, false, b.v, (short)0, acc3[nt], false, false);
    }
  }
  #pragma unroll
  for (int nt = 0; nt < 2; ++nt) {
    int n = nt * 16 + col;
    float bias = b2[n];
    #pragma unroll
    for (int r = 0; r < 8; ++r) {
      float x = acc3[nt][r] + bias;
      sH3[(rowBase + r) * 32 + n] = swishf(x);
    }
  }

  // ---- folded epilogue: out = h3 @ Wc^T + bc   (Wc = lb@la) ----
  if (lane < 16) {
    int row = m0 + lane;
    const float* hr = sH3 + row * 32;
    float o0 = wsBc[0], o1 = wsBc[1], o2 = wsBc[2];
    #pragma unroll
    for (int k = 0; k < 32; ++k) {
      float v = hr[k];
      o0 += v * wsWc[k];
      o1 += v * wsWc[32 + k];
      o2 += v * wsWc[64 + k];
    }
    float* op = out + (long)(p0 + row) * 3;
    op[0] = o0; op[1] = o1; op[2] = o2;
  }
}

// ---------------------------------------------------------------------------
extern "C" void kernel_launch(void* const* d_in, const int* in_sizes, int n_in,
                              void* d_out, int out_size, void* d_ws, size_t ws_size,
                              hipStream_t stream) {
  const float* node_attr = (const float*)d_in[0];
  const float* t         = (const float*)d_in[1];
  const float* perturb   = (const float*)d_in[2];
  const float* Wf        = (const float*)d_in[3];
  const float* embW      = (const float*)d_in[4];
  const float* embB      = (const float*)d_in[5];
  const float* upW       = (const float*)d_in[6];
  const float* upB       = (const float*)d_in[7];
  const float* l0W       = (const float*)d_in[8];
  const float* l0b       = (const float*)d_in[9];
  const float* l1W       = (const float*)d_in[10];
  const float* l1b       = (const float*)d_in[11];
  const float* l2W       = (const float*)d_in[12];
  const float* l2b       = (const float*)d_in[13];
  const float* laW       = (const float*)d_in[14];
  const float* lab       = (const float*)d_in[15];
  const float* lbW       = (const float*)d_in[16];
  const float* lbb       = (const float*)d_in[17];
  const int*   paths     = (const int*)d_in[18];
  const int*   nump      = (const int*)d_in[19];

  const int P = in_sizes[2] / 3;   // perturb_directions is [P,3]
  const int B = in_sizes[1];       // t is [B]

  char* ws = (char*)d_ws;
  float*          wsEmb    = (float*)ws;                      // 64*80 f32   = 20480 B
  unsigned short* wsWbf    = (unsigned short*)(ws + 20480);   // 43008 bf16  = 86016 B
  int*            wsPrefix = (int*)(ws + 106496);             // B+1 ints
  float*          wsWc     = (float*)(ws + 106816);           // 3*32 f32
  float*          wsBc     = (float*)(ws + 107200);           // 3 f32

  prep_kernel<<<1, 256, 0, stream>>>(t, Wf, embW, embB, l0W, l1W, l2W,
                                     laW, lab, lbW, lbb, nump,
                                     wsEmb, wsWbf, wsPrefix, wsWc, wsBc, B);

  const size_t lds = (size_t)(43008 + 32768) * sizeof(unsigned short); // 151552 B
  (void)hipFuncSetAttribute((const void*)mlp_kernel,
                            hipFuncAttributeMaxDynamicSharedMemorySize, (int)lds);
  mlp_kernel<<<P / 128, 256, lds, stream>>>(
      node_attr, perturb, paths, upW, upB, l0b, l1b, l2b,
      wsEmb, wsWbf, wsPrefix, wsWc, wsBc, (float*)d_out, P, B);
}
